// Model_47596827574867
// MI455X (gfx1250) — matrix operand order repaired and checked by reading, not compiled, not run on hardware
//
#include <hip/hip_runtime.h>
#include <hip/hip_bf16.h>

typedef __bf16 bf16_t;
typedef __attribute__((ext_vector_type(16))) __bf16 v16bf;
typedef __attribute__((ext_vector_type(8)))  float  v8f;

#define OUT_CAPS 10
#define IN_CAPS  36
#define OUT_LEN  16
#define IN_LEN   8
#define KDIM     288   // IN_CAPS * IN_LEN
#define NCHUNK   9     // KDIM / 32
#define BT       16    // batch tile (WMMA N)
#define NITER    3
#define NTHREADS 320   // 10 waves, one per out-capsule
#define WB_ELEMS (OUT_CAPS * OUT_LEN * KDIM)   // 46080 per layout

// ---------------------------------------------------------------------------
// Kernel A: lin_w [10][36][16][8] f32 -> two bf16 layouts:
//   wb[o][d][k=i*8+l]  : A-fragments of the out-phase GEMM (contig 16B runs)
//   wt[o][k][d]        : A'-fragments of the agreement GEMM (contig 16B run)
// ---------------------------------------------------------------------------
__global__ void convert_w_kernel(const float* __restrict__ lin_w,
                                 bf16_t* __restrict__ wb,
                                 bf16_t* __restrict__ wt) {
    int idx = blockIdx.x * blockDim.x + threadIdx.x;
    if (idx >= WB_ELEMS) return;
    int o = idx / (OUT_LEN * KDIM);
    int r = idx % (OUT_LEN * KDIM);
    int d = r / KDIM;
    int k = r % KDIM;
    int i = k >> 3, l = k & 7;
    bf16_t v = (bf16_t)lin_w[(((o * IN_CAPS + i) * OUT_LEN) + d) * IN_LEN + l];
    wb[idx] = v;
    wt[(o * KDIM + k) * OUT_LEN + d] = v;
}

// ---------------------------------------------------------------------------
// Kernel B: fused conv+squash -> caps -> routing with WMMA per batch tile.
// ---------------------------------------------------------------------------
__launch_bounds__(NTHREADS)
__global__ void capsnet_kernel(const float* __restrict__ x,
                               const float* __restrict__ conv_w,
                               const bf16_t* __restrict__ wb,
                               const bf16_t* __restrict__ wt,
                               float* __restrict__ out) {
    __shared__ float  s_x[BT * 9];
    __shared__ float  s_cw[32 * 9];
    __shared__ bf16_t s_caps[BT * KDIM];                    // [b][k] bf16
    __shared__ float  s_logits[OUT_CAPS * IN_CAPS * BT];    // [o][i][b]
    __shared__ float  s_probs [OUT_CAPS * IN_CAPS * BT];    // [o][i][b]

    const int tid  = threadIdx.x;
    const int tile = blockIdx.x;
    const int wave = tid >> 5;
    const int lane = tid & 31;
    const int half = lane >> 4;   // 0: lanes 0-15, 1: lanes 16-31
    const int bl   = lane & 15;   // batch-in-tile == WMMA N / M(row) index

    // ---- stage inputs, zero logits ----
    if (tid < BT * 9) s_x[tid]  = x[tile * BT * 9 + tid];
    if (tid < 288)    s_cw[tid] = conv_w[tid];
    for (int j = tid; j < OUT_CAPS * IN_CAPS * BT; j += NTHREADS)
        s_logits[j] = 0.0f;
    __syncthreads();

    // ---- conv (3x3, pad 1, 3x3 spatial) + squash over capsule length 8 ----
    for (int job = tid; job < BT * IN_CAPS; job += NTHREADS) {
        int b = job / IN_CAPS, i = job % IN_CAPS;
        int hw = i >> 2, g = i & 3;
        int h = hw / 3, w = hw % 3;
        float vals[IN_LEN];
        float n2 = 0.0f;
        for (int l = 0; l < IN_LEN; ++l) {
            int c = g * IN_LEN + l;
            float acc = 0.0f;
            for (int kh = 0; kh < 3; ++kh) {
                int r = h + kh - 1;
                if (r < 0 || r > 2) continue;
                for (int kw = 0; kw < 3; ++kw) {
                    int cc = w + kw - 1;
                    if (cc < 0 || cc > 2) continue;
                    acc += s_x[b * 9 + r * 3 + cc] * s_cw[c * 9 + kh * 3 + kw];
                }
            }
            vals[l] = acc;
            n2 += acc * acc;
        }
        float scale = n2 * __builtin_amdgcn_rcpf(1.0f + n2) *
                      __builtin_amdgcn_rsqf(n2 + 1e-8f);
        for (int l = 0; l < IN_LEN; ++l)
            s_caps[b * KDIM + i * IN_LEN + l] = (bf16_t)(vals[l] * scale);
    }
    __syncthreads();

    const int o = wave;                               // one wave per out-capsule
    const bf16_t* wrow_base = wb + o * OUT_LEN * KDIM;

    // ---- preload ALL out-phase A-fragments (iteration-invariant weights) ----
    // A 16x32, this code's contiguous k order: lanes0-15 K=0..15; lanes16-31 K=16..31;
    // row M = lane%16. Two contiguous 16B runs per lane per chunk.
    v16bf afr[NCHUNK];
#pragma unroll
    for (int c = 0; c < NCHUNK; ++c) {
        const bf16_t* wrow = wrow_base + bl * KDIM + c * 32;
#pragma unroll
        for (int e = 0; e < 8; ++e) afr[c][e]     = wrow[16 * half + e];
#pragma unroll
        for (int e = 0; e < 8; ++e) afr[c][8 + e] = wrow[16 * half + 8 + e];
    }
    // common base for agreement A'-fragments: WT[o][k=bl][d=8*half..]
    const bf16_t* wt_base = wt + (o * KDIM + bl) * OUT_LEN + 8 * half;

    for (int it = 0; it < NITER; ++it) {
        // ---- probs = softmax over o (cooperative, all 320 threads) ----
        for (int p = tid; p < IN_CAPS * BT; p += NTHREADS) {
            int b = p & 15, i = p >> 4;
            float lv[OUT_CAPS], m = -1e30f;
#pragma unroll
            for (int oo = 0; oo < OUT_CAPS; ++oo) {
                lv[oo] = s_logits[(oo * IN_CAPS + i) * BT + b];
                m = fmaxf(m, lv[oo]);
            }
            float s = 0.0f;
#pragma unroll
            for (int oo = 0; oo < OUT_CAPS; ++oo) { lv[oo] = __expf(lv[oo] - m); s += lv[oo]; }
            float inv = __builtin_amdgcn_rcpf(s);
#pragma unroll
            for (int oo = 0; oo < OUT_CAPS; ++oo)
                s_probs[(oo * IN_CAPS + i) * BT + b] = lv[oo] * inv;
        }
        __syncthreads();

        // ---- out_unsq[d][b] = sum_k W[o][d][k] * (probs[b,o,k/8] * caps[b][k])
        //      16x16xK GEMM, K=288 = 9 chunks of v_wmma_f32_16x16x32_bf16 ----
        v8f acc = {0.f,0.f,0.f,0.f,0.f,0.f,0.f,0.f};
#pragma unroll
        for (int c = 0; c < NCHUNK; ++c) {
            // B fragment (B 32x16, same contiguous k order as A: lanes0-15 K=0..15,
            //             lanes16-31 K=16..31; col N = lane%16)
            v16bf bfrag;
            int k0 = c * 32 + 16 * half;              // multiple of 8 -> 2 capsules
            const bf16_t* crow = s_caps + bl * KDIM + k0;
            float p0 = s_probs[(o * IN_CAPS + (k0 >> 3)    ) * BT + bl];
            float p1 = s_probs[(o * IN_CAPS + (k0 >> 3) + 1) * BT + bl];
#pragma unroll
            for (int e = 0; e < 8; ++e) bfrag[e]     = (bf16_t)((float)crow[e]     * p0);
#pragma unroll
            for (int e = 0; e < 8; ++e) bfrag[8 + e] = (bf16_t)((float)crow[8 + e] * p1);
            acc = __builtin_amdgcn_wmma_f32_16x16x32_bf16(false, afr[c], false, bfrag,
                                                          (short)0, acc, false, false);
        }

        // ---- squash over d=16: lane holds d = r + 8*half, partner via shfl_xor 16 ----
        float ss = 0.0f;
#pragma unroll
        for (int r = 0; r < 8; ++r) ss += acc[r] * acc[r];
        ss += __shfl_xor(ss, 16, 32);
        float scale = ss * __builtin_amdgcn_rcpf(1.0f + ss) *
                      __builtin_amdgcn_rsqf(ss + 1e-8f);
#pragma unroll
        for (int r = 0; r < 8; ++r) acc[r] *= scale;

        if (it == NITER - 1) {
            // final out[b,o,d], d = 8*half + r contiguous -> two float4 stores
            int bg = tile * BT + bl;
            float* dst = out + (bg * OUT_CAPS + o) * OUT_LEN + 8 * half;
            *(float4*)(dst)     = make_float4(acc[0], acc[1], acc[2], acc[3]);
            *(float4*)(dst + 4) = make_float4(acc[4], acc[5], acc[6], acc[7]);
        } else {
            // ---- agreement: logits[b,o,i] += sum_l caps[b,i,l] * (W^T out)[i,l,b]
            // gather full out[d=0..15][b] per lane0-15 via partner shuffle
            float other[8];
#pragma unroll
            for (int r = 0; r < 8; ++r) other[r] = __shfl_xor(acc[r], 16, 32);
            v16bf bout;   // B = out (K=d, 16 real + 16 zero-pad)
#pragma unroll
            for (int e = 0; e < 16; ++e) {
                float v = (e < 8) ? acc[e] : 0.0f;
                bout[e] = (bf16_t)v;
            }
#pragma unroll 3
            for (int mc = 0; mc < IN_CAPS / 2; ++mc) {
                // A' rows m=(i,l): k = mc*16 + m ; K(e<8) = 8*half + e = d,
                // K(e>=8) >= 16 -> zero pad. One contiguous 16B load per lane:
                // WT[o][mc*16 + bl][8*half + 0..7]  (= wt_base + mc*256 elems)
                v16bf ap;
                const bf16_t* wtrow = wt_base + mc * 16 * OUT_LEN;
#pragma unroll
                for (int e = 0; e < 8; ++e) ap[e]     = wtrow[e];
#pragma unroll
                for (int e = 0; e < 8; ++e) ap[8 + e] = (bf16_t)0.0f;
                v8f vz = {0.f,0.f,0.f,0.f,0.f,0.f,0.f,0.f};
                v8f V = __builtin_amdgcn_wmma_f32_16x16x32_bf16(false, ap, false, bout,
                                                                (short)0, vz, false, false);
                // D row M = r + 8*half -> (i = 2*mc + half, l = r)
                int i = 2 * mc + half;
                const bf16_t* crow = s_caps + bl * KDIM + i * IN_LEN;
                float agr = 0.0f;
#pragma unroll
                for (int r = 0; r < 8; ++r) agr += V[r] * (float)crow[r];
                s_logits[(o * IN_CAPS + i) * BT + bl] += agr;
            }
        }
        __syncthreads();
    }
}

// ---------------------------------------------------------------------------
extern "C" void kernel_launch(void* const* d_in, const int* in_sizes, int n_in,
                              void* d_out, int out_size, void* d_ws, size_t ws_size,
                              hipStream_t stream) {
    const float* x      = (const float*)d_in[0];   // [B,1,3,3]
    const float* conv_w = (const float*)d_in[1];   // [32,1,3,3]
    const float* lin_w  = (const float*)d_in[2];   // [10,36,16,8]
    float* out = (float*)d_out;                    // [B,10,16] f32
    bf16_t* wb = (bf16_t*)d_ws;                    // [o][d][k] bf16, 92160 B
    bf16_t* wt = wb + WB_ELEMS;                    // [o][k][d] bf16, 92160 B

    int B = in_sizes[0] / 9;
    convert_w_kernel<<<(WB_ELEMS + 255) / 256, 256, 0, stream>>>(lin_w, wb, wt);
    capsnet_kernel<<<B / BT, NTHREADS, 0, stream>>>(x, conv_w, wb, wt, out);
}
